// InteractionNetwork_42502996362059
// MI455X (gfx1250) — matrix operand order repaired and checked by reading, not compiled, not run on hardware
//
#include <hip/hip_runtime.h>

typedef unsigned short u16;
typedef unsigned int   u32;
typedef unsigned long long u64;

typedef __attribute__((ext_vector_type(16))) __bf16 v16bf;
typedef __attribute__((ext_vector_type(8)))  float  v8f;
typedef __attribute__((ext_vector_type(4)))  u32    v4u;

union FragBF { v16bf v; u32 u[8]; v4u q[2]; };

// ---------------- constants ----------------
#define NB     128                  // batch
#define NN     64                   // objects per batch
#define MEDG   4032                 // N*(N-1) edges per batch
#define ROWS   (NB*MEDG)            // 516096 total edge rows
#define TM     128                  // edge rows per workgroup tile
#define K0P    64                   // layer0 K padded (37+bias -> 64)
#define HP     160                  // hidden padded (150+bias -> 160)
#define OP     64                   // output padded (50 -> 64)
#define OBJF   17                   // object feature dim
#define HREAL  150                  // real hidden width (bias column index)

__device__ __forceinline__ u16 f2bf(float f) {
    __bf16 h = (__bf16)f;           // hardware RNE convert
    union { __bf16 h; u16 u; } v; v.h = h;
    return v.u;
}

__device__ __forceinline__ float relu1(float x) {
    // single v_med3_f32: clamp to [0, +inf)
    return __builtin_amdgcn_fmed3f(x, 0.0f, __builtin_inff());
}

// ---- CDNA5 async global->LDS staging (ASYNCcnt-tracked, no VGPR round trip) ----
__device__ __forceinline__ void async_stage_b128(const u32* __restrict__ g,
                                                 u32* lds, int n_u32, int tid) {
    u32 lbase = (u32)(size_t)lds;
    u64 gbase = (u64)(size_t)g;
    for (int i = tid; i < (n_u32 >> 2); i += 256) {
        u32 loff = lbase + ((u32)i << 4);
        u64 ga   = gbase + ((u64)(u32)i << 4);
        asm volatile("global_load_async_to_lds_b128 %0, %1, off"
                     :: "v"(loff), "v"(ga) : "memory");
    }
}
__device__ __forceinline__ void wait_async() {
    asm volatile("s_wait_asynccnt 0x0" ::: "memory");
}

// ---------------- prep kernels ----------------
__global__ void prep_objects(const float* __restrict__ tensor, float* __restrict__ objects) {
    int g = blockIdx.x * blockDim.x + threadIdx.x;
    if (g >= NB * NN) return;
    int b = g >> 6, n = g & 63;
    const float* t1 = tensor + (((size_t)b * 2 + 1) * NN + n) * 14;
    const float* t0 = tensor + (((size_t)b * 2 + 0) * NN + n) * 14;
    float* o = objects + (size_t)g * OBJF;
    o[0] = t1[0]; o[1] = t1[1]; o[2] = t1[2];
    o[3] = t0[0]; o[4] = t0[1]; o[5] = t0[2];
    #pragma unroll
    for (int i = 0; i < 11; ++i) o[6 + i] = t1[3 + i];
}

__global__ void zero_agg(float* __restrict__ agg, int count) {
    int g = blockIdx.x * blockDim.x + threadIdx.x;
    if (g < count) agg[g] = 0.0f;
}

// pack fp32 weight [Kin x Nout] + bias -> bf16 pair-interleaved u32, TRANSPOSED:
// dst[n * (Kp/2) + kp] = { bf16(val(2kp,n)), bf16(val(2kp+1,n)) }
// val(k,n) = W[k][n] for k<Kin; bias[n] at k==Kin (augmented GEMM); 0 otherwise.
__global__ void pack_w(const float* __restrict__ src, const float* __restrict__ bias,
                       u32* __restrict__ dst, int Kin, int Nout, int Kp, int Np) {
    int idx = blockIdx.x * blockDim.x + threadIdx.x;
    int kph = Kp >> 1;
    int total = kph * Np;
    if (idx >= total) return;
    int n = idx / kph, kp = idx - n * kph;
    int k0 = kp * 2, k1 = k0 + 1;
    float f0 = 0.0f, f1 = 0.0f;
    if (n < Nout) {
        if (k0 < Kin) f0 = src[k0 * Nout + n]; else if (k0 == Kin) f0 = bias[n];
        if (k1 < Kin) f1 = src[k1 * Nout + n]; else if (k1 == Kin) f1 = bias[n];
    }
    dst[idx] = (u32)f2bf(f0) | ((u32)f2bf(f1) << 16);
}

// ---------------- fused edge MLP ----------------
// Weights in LDS: u32 pairs, layout [col][kpair], col stride = KB*16 u32.
template <int KB, int CB, bool RELU, bool FINAL>
__device__ __forceinline__ void mlp_layer(const u16* __restrict__ inL, int inStride,
                                          const u32* __restrict__ wL,
                                          u16* __restrict__ outL, int outStride,
                                          float* __restrict__ agg, int row0, int tid) {
    constexpr int KPH = KB * 16;         // K-pairs per column
    const int lane  = tid & 31;
    const int wv    = tid >> 5;          // wave id -> row block
    const int r0    = wv * 16;
    const int rowA  = r0 + (lane & 15);
    const int khalf = lane >> 4;
    const int col0  = lane & 15;

    v8f acc[CB];
    #pragma unroll
    for (int cb = 0; cb < CB; ++cb) {
        #pragma unroll
        for (int i = 0; i < 8; ++i) acc[cb][i] = 0.0f;   // folds to inline-0 SRC2
    }

    #pragma unroll
    for (int kb = 0; kb < KB; ++kb) {
        // A 16x32 bf16 fragment: two contiguous 16B groups per lane
        FragBF a;
        const u16* ap = &inL[rowA * inStride + kb * 32 + (khalf << 4)];
        a.q[0] = *(const v4u*)(ap);        // K = kb*32 + khalf*16 + 0..7
        a.q[1] = *(const v4u*)(ap + 8);    // K = kb*32 + khalf*16 + 8..15
        #pragma unroll
        for (int cb = 0; cb < CB; ++cb) {
            // B 32x16 bf16 fragment: 8 consecutive kpairs for this column
            FragBF bm;
            const u32* bp = &wL[(cb * 16 + col0) * KPH + kb * 16 + (khalf << 3)];
            bm.q[0] = *(const v4u*)(bp);
            bm.q[1] = *(const v4u*)(bp + 4);
            acc[cb] = __builtin_amdgcn_wmma_f32_16x16x32_bf16(
                false, a.v, false, bm.v, (short)0, acc[cb], false, false);
        }
    }

    // epilogue: C/D layout -> row = r0 + v + 8*khalf, col = cb*16 + col0
    if constexpr (!FINAL) {
        #pragma unroll
        for (int cb = 0; cb < CB; ++cb) {
            #pragma unroll
            for (int v = 0; v < 8; ++v) {
                int r = r0 + v + 8 * khalf;
                int c = cb * 16 + col0;
                float x = acc[cb][v];
                if constexpr (RELU) x = relu1(x);
                // bias column of the NEXT layer's augmented input carries 1.0
                outL[r * outStride + c] = (c == HREAL) ? (u16)0x3F80 : f2bf(x);
            }
        }
    } else {
        #pragma unroll
        for (int v = 0; v < 8; ++v) {
            int r = r0 + v + 8 * khalf;
            int grow = row0 + r;
            int b = grow / MEDG;
            int e = grow - b * MEDG;
            int recv = e / 63;
            float* dst = &agg[((size_t)(b * NN + recv)) * OP];
            #pragma unroll
            for (int cb = 0; cb < CB; ++cb) {
                int c = cb * 16 + col0;
                if (c < 50) atomicAdd(dst + c, acc[cb][v]);
            }
        }
    }
}

__global__ __launch_bounds__(256) void edge_mlp(
        const float* __restrict__ objects,
        const u32* __restrict__ w0p, const u32* __restrict__ w1p,
        const u32* __restrict__ w2p, const u32* __restrict__ w3p,
        const u32* __restrict__ w4p,
        float* __restrict__ agg) {
    __shared__ alignas(16) u16  act0[TM * HP];
    __shared__ alignas(16) u16  act1[TM * HP];
    __shared__ alignas(16) u32  wl[(HP / 2) * HP];

    const int tid  = threadIdx.x;
    const int row0 = blockIdx.x * TM;

    // layer-0 weights: async DMA straight into LDS while we stage activations
    async_stage_b128(w0p, wl, (K0P / 2) * HP, tid);

    // stage marshalled edge features (37 real + bias col at 37, padded to 64) as bf16
    {
        int r = tid >> 1, half = tid & 1;
        int grow = row0 + r;
        int b = grow / MEDG;
        int e = grow - b * MEDG;
        int recv = e / 63;
        int j = e - recv * 63;
        int send = j + (j >= recv ? 1 : 0);
        const float* objR = objects + ((size_t)(b * NN + recv)) * OBJF;
        const float* objS = objects + ((size_t)(b * NN + send)) * OBJF;
        for (int c = half * 32; c < half * 32 + 32; ++c) {
            float f;
            if (c < 17)       f = objR[c];
            else if (c < 34)  f = objS[c - 17];
            else if (c == 34) f = 1.0f;   // rel[:,0] = 1
            else if (c == 37) f = 1.0f;   // augmented bias column
            else              f = 0.0f;
            act0[r * K0P + c] = f2bf(f);
        }
    }

    // ---- layer 0: 37+1(->64) -> 150(->160) ----
    wait_async();
    __syncthreads();
    mlp_layer<2, 10, true, false>(act0, K0P, wl, act1, HP, nullptr, row0, tid);
    __syncthreads();

    // ---- layer 1 ----
    async_stage_b128(w1p, wl, (HP / 2) * HP, tid);
    wait_async();
    __syncthreads();
    mlp_layer<5, 10, true, false>(act1, HP, wl, act0, HP, nullptr, row0, tid);
    __syncthreads();

    // ---- layer 2 ----
    async_stage_b128(w2p, wl, (HP / 2) * HP, tid);
    wait_async();
    __syncthreads();
    mlp_layer<5, 10, true, false>(act0, HP, wl, act1, HP, nullptr, row0, tid);
    __syncthreads();

    // ---- layer 3 ----
    async_stage_b128(w3p, wl, (HP / 2) * HP, tid);
    wait_async();
    __syncthreads();
    mlp_layer<5, 10, true, false>(act1, HP, wl, act0, HP, nullptr, row0, tid);
    __syncthreads();

    // ---- layer 4: 150+1(->160) -> 50(->64), no relu, atomic scatter ----
    async_stage_b128(w4p, wl, (HP / 2) * OP, tid);
    wait_async();
    __syncthreads();
    mlp_layer<5, 4, false, true>(act0, HP, wl, nullptr, 0, agg, row0, tid);
}

// ---------------- node MLP + transpose ----------------
__global__ __launch_bounds__(64) void node_mlp(
        const float* __restrict__ objects, const float* __restrict__ agg,
        const float* __restrict__ sw0, const float* __restrict__ sb0,
        const float* __restrict__ sw1, const float* __restrict__ sb1,
        float* __restrict__ out) {
    __shared__ float w0[68 * 100];
    __shared__ float b0[100];
    __shared__ float w1[100 * 3];
    __shared__ float b1[4];
    __shared__ float feat[68][64];
    __shared__ float hid[100][64];

    int tid = threadIdx.x;
    for (int i = tid; i < 68 * 100; i += 64) w0[i] = sw0[i];
    for (int i = tid; i < 100; i += 64) b0[i] = sb0[i];
    for (int i = tid; i < 300; i += 64) w1[i] = sw1[i];
    if (tid < 3) b1[tid] = sb1[tid];

    int g = blockIdx.x * 64 + tid;
    int b = g >> 6, n = g & 63;
    for (int c = 0; c < 17; ++c) feat[c][tid] = objects[((size_t)g) * OBJF + c];
    feat[17][tid] = 1.0f;
    for (int c = 0; c < 50; ++c) feat[18 + c][tid] = agg[((size_t)g) * OP + c];
    __syncthreads();

    for (int h = 0; h < 100; ++h) {
        float s = b0[h];
        for (int j = 0; j < 68; ++j) s += feat[j][tid] * w0[j * 100 + h];
        hid[h][tid] = s > 0.0f ? s : 0.0f;
    }
    for (int c = 0; c < 3; ++c) {
        float s = b1[c];
        for (int h = 0; h < 100; ++h) s += hid[h][tid] * w1[h * 3 + c];
        out[((size_t)(b * 3 + c)) * NN + n] = s;   // states.transpose(0,2,1)
    }
}

// ---------------- launch ----------------
extern "C" void kernel_launch(void* const* d_in, const int* in_sizes, int n_in,
                              void* d_out, int out_size, void* d_ws, size_t ws_size,
                              hipStream_t stream) {
    const float* tensor = (const float*)d_in[0];
    const float* iw0 = (const float*)d_in[1];  const float* ib0 = (const float*)d_in[2];
    const float* iw1 = (const float*)d_in[3];  const float* ib1 = (const float*)d_in[4];
    const float* iw2 = (const float*)d_in[5];  const float* ib2 = (const float*)d_in[6];
    const float* iw3 = (const float*)d_in[7];  const float* ib3 = (const float*)d_in[8];
    const float* iw4 = (const float*)d_in[9];  const float* ib4 = (const float*)d_in[10];
    const float* sw0 = (const float*)d_in[11]; const float* sb0 = (const float*)d_in[12];
    const float* sw1 = (const float*)d_in[13]; const float* sb1 = (const float*)d_in[14];
    float* out = (float*)d_out;

    // workspace layout
    float* objects = (float*)d_ws;                        // 128*64*17
    float* agg     = objects + NB * NN * OBJF;            // 128*64*64
    u32*   w0p     = (u32*)(agg + NB * NN * OP);          // 160 cols * 32 kpairs
    u32*   w1p     = w0p + (K0P / 2) * HP;                // 160 cols * 80 kpairs
    u32*   w2p     = w1p + (HP / 2) * HP;
    u32*   w3p     = w2p + (HP / 2) * HP;
    u32*   w4p     = w3p + (HP / 2) * HP;                 // 64 cols * 80 kpairs

    prep_objects<<<(NB * NN + 255) / 256, 256, 0, stream>>>(tensor, objects);
    zero_agg<<<(NB * NN * OP + 255) / 256, 256, 0, stream>>>(agg, NB * NN * OP);

    pack_w<<<((K0P / 2) * HP + 255) / 256, 256, 0, stream>>>(iw0, ib0, w0p, 37, 150, K0P, HP);
    pack_w<<<((HP  / 2) * HP + 255) / 256, 256, 0, stream>>>(iw1, ib1, w1p, 150, 150, HP, HP);
    pack_w<<<((HP  / 2) * HP + 255) / 256, 256, 0, stream>>>(iw2, ib2, w2p, 150, 150, HP, HP);
    pack_w<<<((HP  / 2) * HP + 255) / 256, 256, 0, stream>>>(iw3, ib3, w3p, 150, 150, HP, HP);
    pack_w<<<((HP  / 2) * OP + 255) / 256, 256, 0, stream>>>(iw4, ib4, w4p, 150, 50, HP, OP);

    edge_mlp<<<ROWS / TM, 256, 0, stream>>>(objects, w0p, w1p, w2p, w3p, w4p, agg);

    node_mlp<<<NB * NN / 64, 64, 0, stream>>>(objects, agg, sw0, sb0, sw1, sb1, out);
}
